// AdaptiveFusion_38749194944882
// MI455X (gfx1250) — matrix operand order repaired and checked by reading, not compiled, not run on hardware
//
#include <hip/hip_runtime.h>

// ---- CDNA5 (gfx1250) types -------------------------------------------------
typedef __attribute__((ext_vector_type(16))) __bf16 v16bf;
typedef __attribute__((ext_vector_type(8)))  float  v8f;
typedef __attribute__((ext_vector_type(4)))  float  v4f;

// Problem constants (match reference: V=3, N=131072, D=512)
constexpr int V_VIEWS = 3;
constexpr int NROWS   = 131072;
constexpr int DDIM    = 512;
constexpr int TILE_M  = 16;              // rows per wave tile (WMMA M)
constexpr int WAVES   = 8;               // waves per block (wave32!)
constexpr int BLOCK   = WAVES * 32;      // 256 threads
constexpr float NEG_SLOPE = 0.01f;

// Build the 16x32 bf16 A-fragment for one view from row-major fp32 data.
// Contiguous K layout (16x32): lanes 0-15 hold K[0..15];
// lanes 16-31 hold K[16..31].
// v16bf element e  <->  K = hi*16 + e.
__device__ inline v16bf make_a_frag(const float* __restrict__ p, int k0, int hi) {
    const v4f* q0 = reinterpret_cast<const v4f*>(p + k0 + hi * 16);       // K seg0 (8 f32, 32B)
    const v4f* q1 = reinterpret_cast<const v4f*>(p + k0 + hi * 16 + 8);   // K seg1 (8 f32, 32B)
    v4f f0 = q0[0], f1 = q0[1], f2 = q1[0], f3 = q1[1];
    v16bf a;
    a[0]  = (__bf16)f0.x; a[1]  = (__bf16)f0.y; a[2]  = (__bf16)f0.z; a[3]  = (__bf16)f0.w;
    a[4]  = (__bf16)f1.x; a[5]  = (__bf16)f1.y; a[6]  = (__bf16)f1.z; a[7]  = (__bf16)f1.w;
    a[8]  = (__bf16)f2.x; a[9]  = (__bf16)f2.y; a[10] = (__bf16)f2.z; a[11] = (__bf16)f2.w;
    a[12] = (__bf16)f3.x; a[13] = (__bf16)f3.y; a[14] = (__bf16)f3.z; a[15] = (__bf16)f3.w;
    return a;
}

__global__ __launch_bounds__(BLOCK)
void adaptive_fusion_kernel(const float* __restrict__ xs,
                            const float* __restrict__ alpha_w,
                            const float* __restrict__ alpha_b,
                            float* __restrict__ out) {
    // alpha_w staged once as bf16; the replicated B-fragment (B[k][n]=w[k] for
    // all n) is then a single aligned 32B LDS read per K-chunk per lane group.
    __shared__ __align__(32) __bf16 swb[DDIM];

    const int t    = threadIdx.x;
    const int lane = t & 31;
    const int wave = t >> 5;
    const int hi   = lane >> 4;          // lane group -> K sub-ranges
    const int m    = lane & 15;          // A-operand row within the tile

    swb[t]         = (__bf16)alpha_w[t];
    swb[t + BLOCK] = (__bf16)alpha_w[t + BLOCK];
    __syncthreads();

    const float bval = alpha_b[0];
    const int   row0 = (blockIdx.x * WAVES + wave) * TILE_M;

    // ---------------- Phase 1: logits via WMMA (matrix pipe) ----------------
    const size_t rowOff = (size_t)(row0 + m) * DDIM;
    const float* p0 = xs + rowOff;                                   // view 0
    const float* p1 = xs + (size_t)NROWS * DDIM + rowOff;            // view 1
    const float* p2 = xs + (size_t)2 * NROWS * DDIM + rowOff;        // view 2

    v8f acc0 = {}, acc1 = {}, acc2 = {};
    for (int kc = 0; kc < DDIM / 32; ++kc) {
        const int k0 = kc * 32;
        // Replicated-column B fragment: lanes 0-15 -> K[k0..k0+15],
        // lanes 16-31 -> K[k0+16..k0+31] (identical across N -> broadcast dot).
        const v16bf bfrag = *reinterpret_cast<const v16bf*>(swb + k0 + hi * 16);

        v16bf a0 = make_a_frag(p0, k0, hi);
        acc0 = __builtin_amdgcn_wmma_f32_16x16x32_bf16(false, a0, false, bfrag,
                                                       (short)0, acc0, false, false);
        v16bf a1 = make_a_frag(p1, k0, hi);
        acc1 = __builtin_amdgcn_wmma_f32_16x16x32_bf16(false, a1, false, bfrag,
                                                       (short)0, acc1, false, false);
        v16bf a2 = make_a_frag(p2, k0, hi);
        acc2 = __builtin_amdgcn_wmma_f32_16x16x32_bf16(false, a2, false, bfrag,
                                                       (short)0, acc2, false, false);
    }

    // C layout: VGPR r holds M = r + 8*hi (every column identical here), so
    // each lane owns full dots for 8 rows -> leaky_relu + 3-way softmax in regs.
    float att0[8], att1[8], att2[8];
#pragma unroll
    for (int r = 0; r < 8; ++r) {
        float l0 = acc0[r] + bval; l0 = fmaxf(l0, NEG_SLOPE * l0);
        float l1 = acc1[r] + bval; l1 = fmaxf(l1, NEG_SLOPE * l1);
        float l2 = acc2[r] + bval; l2 = fmaxf(l2, NEG_SLOPE * l2);
        float mx = fmaxf(l0, fmaxf(l1, l2));
        float e0 = __expf(l0 - mx);
        float e1 = __expf(l1 - mx);
        float e2 = __expf(l2 - mx);
        float inv = 1.0f / (e0 + e1 + e2);
        att0[r] = e0 * inv; att1[r] = e1 * inv; att2[r] = e2 * inv;
    }

    // ---------------- Phase 2: fp32 weighted sum (L2-resident re-read) ------
    const float* v0 = xs + (size_t)row0 * DDIM;
    const float* v1 = xs + ((size_t)NROWS + row0) * DDIM;
    const float* v2 = xs + ((size_t)2 * NROWS + row0) * DDIM;
    float* ob = out + (size_t)row0 * DDIM;

#pragma unroll
    for (int r = 0; r < TILE_M; ++r) {
        // rows 0-7 live in lanes 0-15 (take lane 0); rows 8-15 in lanes 16-31.
        const int src = (r < 8) ? 0 : 16;
        const float a0 = __shfl(att0[r & 7], src, 32);   // -> v_readlane
        const float a1 = __shfl(att1[r & 7], src, 32);
        const float a2 = __shfl(att2[r & 7], src, 32);

        const v4f* x0 = reinterpret_cast<const v4f*>(v0 + (size_t)r * DDIM);
        const v4f* x1 = reinterpret_cast<const v4f*>(v1 + (size_t)r * DDIM);
        const v4f* x2 = reinterpret_cast<const v4f*>(v2 + (size_t)r * DDIM);
        v4f* o = reinterpret_cast<v4f*>(ob + (size_t)r * DDIM);

#pragma unroll
        for (int j = 0; j < DDIM / 4 / 32; ++j) {        // 4 coalesced b128 chunks
            const int c = lane + j * 32;
            v4f q0 = x0[c], q1 = x1[c], q2 = x2[c];
            v4f res = a0 * q0 + a1 * q1 + a2 * q2;
            __builtin_nontemporal_store(res, o + c);     // NT: don't pollute L2
        }
    }
}

extern "C" void kernel_launch(void* const* d_in, const int* in_sizes, int n_in,
                              void* d_out, int out_size, void* d_ws, size_t ws_size,
                              hipStream_t stream) {
    (void)in_sizes; (void)n_in; (void)out_size; (void)d_ws; (void)ws_size;
    const float* xs      = (const float*)d_in[0];
    const float* alpha_w = (const float*)d_in[1];
    const float* alpha_b = (const float*)d_in[2];
    float* out = (float*)d_out;

    dim3 grid(NROWS / (TILE_M * WAVES));   // 1024 blocks
    dim3 block(BLOCK);                     // 256 threads = 8 wave32
    hipLaunchKernelGGL(adaptive_fusion_kernel, grid, block, 0, stream,
                       xs, alpha_w, alpha_b, out);
}
